// MultiHeadAttention_69526930587838
// MI455X (gfx1250) — matrix operand order repaired and checked by reading, not compiled, not run on hardware
//
#include <hip/hip_runtime.h>
#include <hip/hip_bf16.h>

// ---------- types ----------
typedef unsigned short u16;
typedef unsigned int   u32;
typedef unsigned long long u64;
typedef __attribute__((ext_vector_type(4)))  u32    u32x4;
typedef __attribute__((ext_vector_type(4)))  int    i32x4;
typedef __attribute__((ext_vector_type(16))) __bf16 v16bf;
typedef __attribute__((ext_vector_type(8)))  float  v8f;

#define AS1 __attribute__((address_space(1)))
#define AS3 __attribute__((address_space(3)))

union FragAB {
  v16bf v;
  u32x4 q[2];
};

__device__ __forceinline__ u16 f2bf(float f) {
  u32 u = __float_as_uint(f);
  u32 r = (u + 0x7FFFu + ((u >> 16) & 1u)) >> 16;   // round-to-nearest-even
  return (u16)r;
}

__device__ __forceinline__ v8f vzero8() {
  v8f v = {0.f, 0.f, 0.f, 0.f, 0.f, 0.f, 0.f, 0.f};
  return v;
}

// ---------- async global->LDS copy (CDNA5 path, with fallbacks) ----------
#if __has_builtin(__builtin_amdgcn_global_load_async_to_lds_b128)
__device__ __forceinline__ void async_copy_b128(const u16* g, u16* l) {
  // signature (from compiler diagnostic): (int4 __device__*, int4 __shared__*, imm, imm)
  __builtin_amdgcn_global_load_async_to_lds_b128(
      (AS1 i32x4*)(u16*)g, (AS3 i32x4*)l, 0, 0);
}
#else
__device__ __forceinline__ void async_copy_b128(const u16* g, u16* l) {
  u32 lofs = (u32)(size_t)(AS3 void*)l;     // LDS byte offset
  u64 gaddr = (u64)(size_t)g;
  asm volatile("global_load_async_to_lds_b128 %0, %1, off"
               :: "v"(lofs), "v"(gaddr) : "memory");
}
#endif

__device__ __forceinline__ void wait_asynccnt0() {
#if __has_builtin(__builtin_amdgcn_s_wait_asynccnt)
  __builtin_amdgcn_s_wait_asynccnt(0);
#else
  asm volatile("s_wait_asynccnt 0" ::: "memory");
#endif
}

#define D_MODEL   1024
#define NUM_HEADS 16
#define HEAD_DIM  64
#define SEQ       2048
#define BATCH     2

// ============================================================================
// fp32 -> bf16 elementwise convert (2 elems / thread, packed u32 store)
// ============================================================================
__global__ __launch_bounds__(256) void cvt_bf16_kernel(const float* __restrict__ in,
                                                       u16* __restrict__ out, int n2) {
  int i = blockIdx.x * 256 + threadIdx.x;
  if (i < n2) {
    u32 lo = f2bf(in[2 * i]);
    u32 hi = f2bf(in[2 * i + 1]);
    ((u32*)out)[i] = lo | (hi << 16);
  }
}

// ============================================================================
// GEMM: C[MxN] = A[MxK] * B[NxK]^T   (bf16 in, fp32 out)
// block tile 128x128, 8 waves (4 along M x 2 along N), wave tile 32x64,
// K-step 32, async global->LDS staging (ASYNCcnt), v_wmma_f32_16x16x32_bf16.
// ============================================================================
__global__ __launch_bounds__(256) void gemm_bf16_kernel(const u16* __restrict__ A,
                                                        const u16* __restrict__ B,
                                                        float* __restrict__ C,
                                                        int M, int N, int K) {
  __shared__ __attribute__((aligned(16))) u16 As[128 * 32];
  __shared__ __attribute__((aligned(16))) u16 Bs[128 * 32];

  const int t    = threadIdx.x;
  const int lane = t & 31;
  const int wave = t >> 5;
  const int wm   = wave & 3;   // 0..3 : 32-row slabs
  const int wn   = wave >> 2;  // 0..1 : 64-col slabs
  const int bm   = blockIdx.y * 128;
  const int bn   = blockIdx.x * 128;

  const int arow = lane & 15;
  const int kkA  = (lane >> 4) * 8;    // A: two 8-elem chunks {kkA, 16+kkA}
  const int kkB  = (lane >> 4) * 16;   // unused: B loads use the kkA chunks too

  v8f acc[2][4];
#pragma unroll
  for (int mi = 0; mi < 2; ++mi)
#pragma unroll
    for (int ni = 0; ni < 4; ++ni) acc[mi][ni] = vzero8();

  for (int k0 = 0; k0 < K; k0 += 32) {
    // --- stage 128x32 tiles of A and B into LDS via async copies ---
#pragma unroll
    for (int j = 0; j < 2; ++j) {
      int idx = t + j * 256;           // 0..511
      int r   = idx >> 2;
      int c   = (idx & 3) * 8;
      async_copy_b128(&A[(size_t)(bm + r) * K + k0 + c], &As[r * 32 + c]);
      async_copy_b128(&B[(size_t)(bn + r) * K + k0 + c], &Bs[r * 32 + c]);
    }
    if (k0 + 32 < K) {   // hint the next K-slice into cache
      __builtin_prefetch(&A[(size_t)(bm + (t >> 1)) * K + k0 + 32], 0, 1);
      __builtin_prefetch(&B[(size_t)(bn + (t >> 1)) * K + k0 + 32], 0, 1);
    }
    wait_asynccnt0();
    __syncthreads();

    FragAB af[2], bf[4];
#pragma unroll
    for (int mi = 0; mi < 2; ++mi) {
      const u16* p = &As[(wm * 32 + mi * 16 + arow) * 32 + kkA];
      af[mi].q[0] = *(const u32x4*)p;
      af[mi].q[1] = *(const u32x4*)(p + 16);
    }
#pragma unroll
    for (int ni = 0; ni < 4; ++ni) {
      const u16* p = &Bs[(wn * 64 + ni * 16 + arow) * 32 + kkA];
      bf[ni].q[0] = *(const u32x4*)p;
      bf[ni].q[1] = *(const u32x4*)(p + 16);
    }
#pragma unroll
    for (int mi = 0; mi < 2; ++mi)
#pragma unroll
      for (int ni = 0; ni < 4; ++ni)
        acc[mi][ni] = __builtin_amdgcn_wmma_f32_16x16x32_bf16(
            false, af[mi].v, false, bf[ni].v, (short)0, acc[mi][ni], false, false);
    __syncthreads();
  }

  // --- write back (C/D layout: lane = {N=l&15, M=r+8*(l>>4)}) ---
  const int half = lane >> 4;
  const int col  = lane & 15;
#pragma unroll
  for (int mi = 0; mi < 2; ++mi)
#pragma unroll
    for (int ni = 0; ni < 4; ++ni)
#pragma unroll
      for (int r = 0; r < 8; ++r) {
        int m = bm + wm * 32 + mi * 16 + r + 8 * half;
        int n = bn + wn * 64 + ni * 16 + col;
        C[(size_t)m * N + n] = acc[mi][ni][r];
      }
}

// ============================================================================
// RoPE + head split: QKV fp32 (M x 3072) -> Qb,Kb (BH,S,64) bf16 with RoPE,
// Vt (BH,64,S) bf16 (transposed so P*V B-operand loads are contiguous).
// One thread per rotation pair.
// ============================================================================
__global__ __launch_bounds__(256) void rope_split_kernel(const float* __restrict__ QKV,
                                                         u16* __restrict__ Qb,
                                                         u16* __restrict__ Kb,
                                                         u16* __restrict__ Vt) {
  int tid  = blockIdx.x * 256 + threadIdx.x;  // BATCH*NUM_HEADS*SEQ*32 threads
  int i    = tid & 31;                        // pair index 0..31
  int rest = tid >> 5;                        // bh*SEQ + s
  int s    = rest & (SEQ - 1);
  int bh   = rest >> 11;                      // SEQ == 2048 == 1<<11
  int h    = bh & (NUM_HEADS - 1);
  int b    = bh >> 4;

  float f  = __powf(10000.0f, -(float)i / 32.0f);
  float ang = (float)s * f;
  float cs, sn;
  __sincosf(ang, &cs, &sn);

  size_t row = ((size_t)(b * SEQ + s)) * (3 * D_MODEL);
  int d = h * HEAD_DIM + 2 * i;
  float q0 = QKV[row + d],                q1 = QKV[row + d + 1];
  float k0 = QKV[row + D_MODEL + d],      k1 = QKV[row + D_MODEL + d + 1];
  float v0 = QKV[row + 2 * D_MODEL + d],  v1 = QKV[row + 2 * D_MODEL + d + 1];

  size_t hq = ((size_t)bh * SEQ + s) * HEAD_DIM + 2 * i;
  Qb[hq]     = f2bf(q0 * cs - q1 * sn);
  Qb[hq + 1] = f2bf(q0 * sn + q1 * cs);
  Kb[hq]     = f2bf(k0 * cs - k1 * sn);
  Kb[hq + 1] = f2bf(k0 * sn + k1 * cs);

  size_t vt = ((size_t)bh * HEAD_DIM + 2 * i) * SEQ + s;
  Vt[vt]       = f2bf(v0);
  Vt[vt + SEQ] = f2bf(v1);
}

// ============================================================================
// Flash attention (causal), per (b,h): block = 8 waves, wave owns 16 q rows.
// Key loop step 32: 4 WMMAs scores (d_k=64) + online softmax + 4 WMMAs P*V.
// P converted acc-layout -> A-layout via per-wave LDS round trip.
// ============================================================================
__global__ __launch_bounds__(256) void flash_kernel(const u16* __restrict__ Qb,
                                                    const u16* __restrict__ Kb,
                                                    const u16* __restrict__ Vt,
                                                    u16* __restrict__ AO) {
  __shared__ __attribute__((aligned(16))) u16 pLds[8][16 * 32];

  const int t    = threadIdx.x;
  const int lane = t & 31;
  const int w    = t >> 5;
  const int bh   = blockIdx.y;               // b*NUM_HEADS + h
  const int qBase = blockIdx.x * 128 + w * 16;

  const u16* Qh = Qb + (size_t)bh * SEQ * HEAD_DIM;
  const u16* Kh = Kb + (size_t)bh * SEQ * HEAD_DIM;
  const u16* Vh = Vt + (size_t)bh * HEAD_DIM * SEQ;

  const int lc   = lane & 15;
  const int half = lane >> 4;
  const int kkA  = half * 8;   // A-operand chunk base
  const int kkB  = half * 16;  // unused: B loads also use kkA

  // Q fragments (A-operand), loaded once: d_k = 64 -> 2 K-steps of 32
  FragAB qf[2];
  {
    const u16* p = Qh + (size_t)(qBase + lc) * HEAD_DIM;
#pragma unroll
    for (int ks = 0; ks < 2; ++ks) {
      qf[ks].q[0] = *(const u32x4*)(p + ks * 32 + kkA);
      qf[ks].q[1] = *(const u32x4*)(p + ks * 32 + kkA + 16);
    }
  }

  float mrow[8], lrow[8];
  v8f oacc[4];
#pragma unroll
  for (int r = 0; r < 8; ++r) { mrow[r] = -1e30f; lrow[r] = 0.f; }
#pragma unroll
  for (int ni = 0; ni < 4; ++ni) oacc[ni] = vzero8();

  const int kEnd = ((qBase + 16 + 31) >> 5) << 5;   // causal bound, uniform per wave

  for (int kb = 0; kb < kEnd; kb += 32) {
    // ---- scores: two 16x16 tiles over 32 keys, K-dim = 64 ----
    v8f sAcc[2];
    sAcc[0] = vzero8();
    sAcc[1] = vzero8();
#pragma unroll
    for (int kc = 0; kc < 2; ++kc)
#pragma unroll
      for (int ks = 0; ks < 2; ++ks) {
        FragAB kf;  // B-operand: lane holds key column (kb+kc*16+lc), two 8-elem d_k chunks {kkA, 16+kkA}
        const u16* p = Kh + (size_t)(kb + kc * 16 + lc) * HEAD_DIM + ks * 32 + kkA;
        kf.q[0] = *(const u32x4*)p;
        kf.q[1] = *(const u32x4*)(p + 16);
        sAcc[kc] = __builtin_amdgcn_wmma_f32_16x16x32_bf16(
            false, qf[ks].v, false, kf.v, (short)0, sAcc[kc], false, false);
      }

    // ---- scale, causal mask, online softmax (rows spread over 16 lanes) ----
#pragma unroll
    for (int r = 0; r < 8; ++r) {
      int q  = qBase + r + 8 * half;
      float v0 = sAcc[0][r] * 0.125f;           // 1/sqrt(64)
      float v1 = sAcc[1][r] * 0.125f;
      if (kb + lc > q)      v0 = -1e30f;
      if (kb + 16 + lc > q) v1 = -1e30f;

      float tm = fmaxf(v0, v1);
      tm = fmaxf(tm, __shfl_xor(tm, 1, 32));
      tm = fmaxf(tm, __shfl_xor(tm, 2, 32));
      tm = fmaxf(tm, __shfl_xor(tm, 4, 32));
      tm = fmaxf(tm, __shfl_xor(tm, 8, 32));

      float nm = fmaxf(mrow[r], tm);
      float sc = __expf(mrow[r] - nm);
      float p0 = __expf(v0 - nm);
      float p1 = __expf(v1 - nm);

      float rs = p0 + p1;
      rs += __shfl_xor(rs, 1, 32);
      rs += __shfl_xor(rs, 2, 32);
      rs += __shfl_xor(rs, 4, 32);
      rs += __shfl_xor(rs, 8, 32);

      lrow[r] = lrow[r] * sc + rs;
      mrow[r] = nm;
#pragma unroll
      for (int ni = 0; ni < 4; ++ni) oacc[ni][r] *= sc;

      int m = r + 8 * half;
      pLds[w][m * 32 + lc]      = f2bf(p0);
      pLds[w][m * 32 + 16 + lc] = f2bf(p1);
    }

    asm volatile("s_wait_dscnt 0" ::: "memory");   // wave-local LDS RAW fence

    // ---- P (16x32) as A-operand from LDS ----
    FragAB pf;
    pf.q[0] = *(const u32x4*)&pLds[w][lc * 32 + kkA];
    pf.q[1] = *(const u32x4*)&pLds[w][lc * 32 + 16 + kkA];

    // ---- O += P * V : V transposed, lane holds d-column, keys in 8-elem chunks {kkA, 16+kkA} ----
#pragma unroll
    for (int ni = 0; ni < 4; ++ni) {
      FragAB vf;
      const u16* vp = Vh + (size_t)(ni * 16 + lc) * SEQ + kb + kkA;
      vf.q[0] = *(const u32x4*)vp;
      vf.q[1] = *(const u32x4*)(vp + 16);
      oacc[ni] = __builtin_amdgcn_wmma_f32_16x16x32_bf16(
          false, pf.v, false, vf.v, (short)0, oacc[ni], false, false);
    }
  }

  // ---- epilogue: normalize, write attn-out bf16 in (B,S,H*64) layout ----
  const int b = bh >> 4;
  const int h = bh & (NUM_HEADS - 1);
#pragma unroll
  for (int r = 0; r < 8; ++r) {
    float inv = 1.0f / lrow[r];
    int q = qBase + r + 8 * half;
    size_t o = ((size_t)(b * SEQ + q)) * D_MODEL + h * HEAD_DIM;
#pragma unroll
    for (int ni = 0; ni < 4; ++ni)
      AO[o + ni * 16 + lc] = f2bf(oacc[ni][r] * inv);
  }
}

// ============================================================================
// host-side orchestration
// ============================================================================
extern "C" void kernel_launch(void* const* d_in, const int* in_sizes, int n_in,
                              void* d_out, int out_size, void* d_ws, size_t ws_size,
                              hipStream_t stream) {
  (void)in_sizes; (void)n_in; (void)out_size; (void)ws_size;

  const float* x    = (const float*)d_in[0];
  const float* Wqkv = (const float*)d_in[1];
  const float* Wo   = (const float*)d_in[2];
  float* out        = (float*)d_out;

  const int M  = BATCH * SEQ;                 // 4096
  const int N1 = 3 * D_MODEL;                 // 3072
  const size_t headElems = (size_t)BATCH * NUM_HEADS * SEQ * HEAD_DIM;  // 4M

  char* ws = (char*)d_ws;
  u16* Xb   = (u16*)ws;   ws += (size_t)M * D_MODEL * sizeof(u16);
  u16* Wqb  = (u16*)ws;   ws += (size_t)N1 * D_MODEL * sizeof(u16);
  u16* Wob  = (u16*)ws;   ws += (size_t)D_MODEL * D_MODEL * sizeof(u16);
  float* QKVf = (float*)ws; ws += (size_t)M * N1 * sizeof(float);
  u16* Qb   = (u16*)ws;   ws += headElems * sizeof(u16);
  u16* Kb   = (u16*)ws;   ws += headElems * sizeof(u16);
  u16* Vt   = (u16*)ws;   ws += headElems * sizeof(u16);
  u16* AOb  = (u16*)ws;   ws += (size_t)M * D_MODEL * sizeof(u16);

  // 1) convert inputs to bf16
  {
    int n2;
    n2 = M * D_MODEL / 2;
    cvt_bf16_kernel<<<(n2 + 255) / 256, 256, 0, stream>>>(x, Xb, n2);
    n2 = N1 * D_MODEL / 2;
    cvt_bf16_kernel<<<(n2 + 255) / 256, 256, 0, stream>>>(Wqkv, Wqb, n2);
    n2 = D_MODEL * D_MODEL / 2;
    cvt_bf16_kernel<<<(n2 + 255) / 256, 256, 0, stream>>>(Wo, Wob, n2);
  }

  // 2) QKV projection: (4096 x 1024) * (3072 x 1024)^T -> fp32
  gemm_bf16_kernel<<<dim3(N1 / 128, M / 128), 256, 0, stream>>>(Xb, Wqb, QKVf,
                                                                M, N1, D_MODEL);

  // 3) RoPE + head split (+ V transpose)
  {
    int threads = BATCH * NUM_HEADS * SEQ * 32;
    rope_split_kernel<<<threads / 256, 256, 0, stream>>>(QKVf, Qb, Kb, Vt);
  }

  // 4) causal flash attention
  flash_kernel<<<dim3(SEQ / 128, BATCH * NUM_HEADS), 256, 0, stream>>>(Qb, Kb, Vt, AOb);

  // 5) output projection: (4096 x 1024) * (1024 x 1024)^T -> fp32 out
  gemm_bf16_kernel<<<dim3(D_MODEL / 128, M / 128), 256, 0, stream>>>(AOb, Wob, out,
                                                                     M, D_MODEL, D_MODEL);
}
